// MetricalChordEncoder_82308753260721
// MI455X (gfx1250) — matrix operand order repaired and checked by reading, not compiled, not run on hardware
//
#include <hip/hip_runtime.h>
#include <math.h>

// ---------------------------------------------------------------------------
// MetricalChordEncoder for MI455X (gfx1250, wave32, WMMA).
//
// Input pointer order assumption (documented): top-level setup_inputs() dict in
// insertion order [x, edge_index, onset_index, onset_idx, lengths, params],
// with `params` flattened in jax pytree order (dict keys sorted, ASCII):
//   5:Wnbr0 6:Wnbr1 7:Wnbr2 8:Wroot0 9:Wroot1 10:Wroot2 11:b0 12:b1 13:b2
//   14:bn1_b 15:bn1_g 16:bn2_b 17:bn2_g 18:emb_W 19:emb_b
//   20..35: gru leaves (l0d0 Wih,Whh,bih,bhh; l0d1; l1d0; l1d1)
//   36:ln_b 37:ln_g 38:pitch_emb 39:proj1_W 40:proj1_b 41:proj2_W 42:proj2_b
//   43:spell_emb 44:trans_W 45:trans_b
// ---------------------------------------------------------------------------

#define N_NODES 100000
#define E_TYPE  400000
#define E_ON    400000
#define N_POOL  50000
#define BATCH   100
#define SEQL    (N_POOL / BATCH)   // 500
#define IN_F    35
#define NH      256
#define HG      128
#define NT      7
#define KP1     288                // proj1 K padded from 257 to 288

typedef __attribute__((ext_vector_type(16))) __bf16 v16bf;
typedef __attribute__((ext_vector_type(8)))  float  v8f;

// ---------------------------------------------------------------------------
// bf16 WMMA GEMM: C[M,NO] += (rowScale ? diag(rowScale) : I) * A[M,lda] @ W[K,NO]
// One wave computes a 16x64 strip of C (4 WMMA accumulators), K in steps of 32.
// Requirements: M % 16 == 0, K % 32 == 0, NO % 64 == 0, A rows 32B-aligned
// (lda % 8 == 0). NO is a template constant so B-fragment loads use immediate
// offsets. Fragment layouts per CDNA5 ISA 7.12.2 (wave32):
//   A 16x32: lane holds row M=lane%16; K offsets are contiguous: half*8+e (e<8)
//            and 16+half*8+(e-8) (e>=8)  -> two aligned float8 loads.
//   B 32x16: b[e] = W[k0 + half*8 + e + (e>=8 ? 8 : 0)][col], col = n-tile + lane%16.
//   C/D:     VGPR r -> row r (+8 for lanes 16-31), col = n-tile + lane%16.
// ---------------------------------------------------------------------------
template <int NO>
__global__ void k_gemm_bf16(const float* __restrict__ A, int lda,
                            const float* __restrict__ rowScale,
                            const float* __restrict__ W,
                            float* __restrict__ C, int M, int K) {
  const int wave = (blockIdx.x * blockDim.x + threadIdx.x) >> 5;
  constexpr int nGroups = NO >> 6;
  const int mt = wave / nGroups;
  if (mt * 16 >= M) return;                 // wave-uniform exit
  const int ng   = wave % nGroups;
  const int lane = threadIdx.x & 31;
  const int half = lane >> 4;               // 0: lanes 0-15, 1: lanes 16-31
  const int ln   = lane & 15;
  const int row  = mt * 16 + ln;            // A-fragment row (both halves)
  const int n0   = ng * 64;
  const float rs = rowScale ? rowScale[row] : 1.0f;
  const float* __restrict__ Arow = A + (size_t)row * lda + half * 8;
  const float* __restrict__ Wb   = W + (size_t)half * 8 * NO + n0 + ln;

  v8f acc[4];
#pragma unroll
  for (int j = 0; j < 4; ++j) acc[j] = (v8f){};

  for (int k0 = 0; k0 < K; k0 += 32) {
    // A fragment: two aligned 32B vector loads + bf16 convert.
    v8f lo = *(const v8f*)(Arow + k0);
    v8f hi = *(const v8f*)(Arow + k0 + 16);
    v16bf a;
#pragma unroll
    for (int e = 0; e < 8; ++e) {
      a[e]     = (__bf16)(lo[e] * rs);
      a[e + 8] = (__bf16)(hi[e] * rs);
    }
    if (k0 + 32 < K)  // prefetch next weight panel -> global_prefetch_b8
      __builtin_prefetch(Wb + (size_t)(k0 + 32) * NO, 0, 1);

    const float* __restrict__ wp = Wb + (size_t)k0 * NO;
#pragma unroll
    for (int j = 0; j < 4; ++j) {
      v16bf b;
#pragma unroll
      for (int e = 0; e < 16; ++e)           // immediate offsets (NO constexpr)
        b[e] = (__bf16)wp[(e + (e >= 8 ? 8 : 0)) * NO + j * 16];
      acc[j] = __builtin_amdgcn_wmma_f32_16x16x32_bf16(
          false, a, false, b, (short)0, acc[j], false, false);
    }
  }
#pragma unroll
  for (int j = 0; j < 4; ++j) {
    const int cc = n0 + j * 16 + ln;
#pragma unroll
    for (int r = 0; r < 8; ++r) {
      const int rr = mt * 16 + r + (half ? 8 : 0);
      C[(size_t)rr * NO + cc] += acc[j][r];
    }
  }
}

// ---------------------------------------------------------------------------
// Elementwise / helper kernels
// ---------------------------------------------------------------------------
__global__ void k_embed(const float* __restrict__ x, const float* __restrict__ embW,
                        const float* __restrict__ embB, const float* __restrict__ pitchE,
                        const float* __restrict__ spellE, float* __restrict__ h) {
  int gid = blockIdx.x * blockDim.x + threadIdx.x;
  int n = gid >> 6, j = gid & 63;
  if (n >= N_NODES) return;
  const float* xr = x + (size_t)n * IN_F;
  float v;
  if (j < 32) {
    float s = embB[j];
#pragma unroll 8
    for (int k = 0; k < 32; ++k) s += xr[2 + k] * embW[k * 32 + j];
    v = s;
  } else if (j < 48) {
    v = pitchE[(int)xr[0] * 16 + (j - 32)];
  } else {
    v = spellE[(int)xr[1] * 16 + (j - 48)];
  }
  h[(size_t)n * 64 + j] = v;
}

__global__ void k_edge_count(const int* __restrict__ ei, float* __restrict__ cnt) {
  int gid = blockIdx.x * blockDim.x + threadIdx.x;
  if (gid >= NT * E_TYPE) return;
  int t = gid / E_TYPE, e = gid - t * E_TYPE;
  int dst = ei[(size_t)t * 2 * E_TYPE + E_TYPE + e];
  atomicAdd(&cnt[(size_t)t * N_NODES + dst], 1.0f);
}

__global__ void k_invert_cnt(float* __restrict__ c, int n) {
  int i = blockIdx.x * blockDim.x + threadIdx.x;
  if (i < n) c[i] = 1.0f / fmaxf(c[i], 1.0f);
}

__global__ void k_scatter4(const int* __restrict__ src, const int* __restrict__ dst,
                           const float* __restrict__ h, float* __restrict__ obuf,
                           int E, int din) {
  const int tpe = din >> 2;
  int gid = blockIdx.x * blockDim.x + threadIdx.x;
  if (gid >= E * tpe) return;
  int e = gid / tpe, c = (gid - e * tpe) << 2;
  int s = src[e], d = dst[e];
  float4 v = *(const float4*)(h + (size_t)s * din + c);
  float* o = obuf + (size_t)d * din + c;
  atomicAdd(o + 0, v.x); atomicAdd(o + 1, v.y);
  atomicAdd(o + 2, v.z); atomicAdd(o + 3, v.w);
}

__global__ void k_sum_types(const float* __restrict__ W, float* __restrict__ out, int sz) {
  int i = blockIdx.x * blockDim.x + threadIdx.x;
  if (i >= sz) return;
  float s = 0.f;
#pragma unroll
  for (int t = 0; t < NT; ++t) s += W[(size_t)t * sz + i];
  out[i] = s;
}

__global__ void k_fill_rowvec(float* __restrict__ C, const float* __restrict__ bias,
                              int M, int NOUT) {
  int i = blockIdx.x * blockDim.x + threadIdx.x;
  if (i < M * NOUT) C[i] = bias[i % NOUT];
}

__global__ void k_fill_const(float* __restrict__ p, float v, int n) {
  int i = blockIdx.x * blockDim.x + threadIdx.x;
  if (i < n) p[i] = v;
}

__global__ void k_scale_relu(const float* __restrict__ a, float* __restrict__ o,
                             int n, float s) {
  int i = blockIdx.x * blockDim.x + threadIdx.x;
  if (i < n) o[i] = fmaxf(a[i] * s, 0.0f);
}

__global__ void k_relu(float* __restrict__ p, int n) {
  int i = blockIdx.x * blockDim.x + threadIdx.x;
  if (i < n) p[i] = fmaxf(p[i], 0.0f);
}

__global__ void k_rownorm(float* __restrict__ h, int M, int D) {
  int wv = (blockIdx.x * blockDim.x + threadIdx.x) >> 5;
  int lane = threadIdx.x & 31;
  if (wv >= M) return;
  float* r = h + (size_t)wv * D;
  float ss = 0.f;
  for (int i = lane; i < D; i += 32) { float v = r[i]; ss += v * v; }
#pragma unroll
  for (int m = 16; m >= 1; m >>= 1) ss += __shfl_xor(ss, m, 32);
  float inv = 1.0f / fmaxf(sqrtf(ss), 1e-12f);
  for (int i = lane; i < D; i += 32) r[i] *= inv;
}

__global__ void k_count_on(const int* __restrict__ dst, float* __restrict__ cnt, int E) {
  int e = blockIdx.x * blockDim.x + threadIdx.x;
  if (e < E) atomicAdd(&cnt[dst[e]], 1.0f);
}

// hp [N_POOL, 288]: cols 0..255 = pooled mean, col 256 = last x feature, rest 0.
__global__ void k_build_hp(const float* __restrict__ pooled, const float* __restrict__ cntOn,
                           const int* __restrict__ idx, const float* __restrict__ x,
                           float* __restrict__ hp) {
  int gid = blockIdx.x * blockDim.x + threadIdx.x;
  int j = gid / KP1, f = gid - j * KP1;
  if (j >= N_POOL) return;
  int g = idx[j];
  float v = 0.0f;
  if (f < 256)       v = pooled[(size_t)g * 256 + f] / cntOn[g];
  else if (f == 256) v = x[(size_t)g * IN_F + (IN_F - 1)];
  hp[(size_t)j * KP1 + f] = v;
}

// proj1_W [257,256] -> padded [288,256] (rows 257..287 zero)
__global__ void k_pad_w1(const float* __restrict__ W, float* __restrict__ Wp) {
  int gid = blockIdx.x * blockDim.x + threadIdx.x;
  if (gid >= KP1 * 256) return;
  Wp[gid] = (gid < 257 * 256) ? W[gid] : 0.0f;
}

__global__ void k_colstats(const float* __restrict__ z, int M, int NOUT,
                           float* __restrict__ stats) {
  int c = blockIdx.x, tid = threadIdx.x;
  float s = 0.f, ss = 0.f;
  for (int j = tid; j < M; j += blockDim.x) {
    float v = z[(size_t)j * NOUT + c];
    s += v; ss += v * v;
  }
  __shared__ float S[256], S2[256];
  S[tid] = s; S2[tid] = ss; __syncthreads();
  for (int step = 128; step > 0; step >>= 1) {
    if (tid < step) { S[tid] += S[tid + step]; S2[tid] += S2[tid + step]; }
    __syncthreads();
  }
  if (tid == 0) { stats[c] = S[0]; stats[NOUT + c] = S2[0]; }
}

__global__ void k_bn_apply(float* __restrict__ z, const float* __restrict__ stats,
                           const float* __restrict__ g, const float* __restrict__ b,
                           int M, int NOUT) {
  int i = blockIdx.x * blockDim.x + threadIdx.x;
  if (i >= M * NOUT) return;
  int c = i % NOUT;
  float m = stats[c] / (float)M;
  float v = stats[NOUT + c] / (float)M - m * m;
  z[i] = g[c] * (z[i] - m) * rsqrtf(v + 1e-5f) + b[c];
}

__global__ void k_transpose(const float* __restrict__ Wih, float* __restrict__ WT, int K) {
  int gid = blockIdx.x * blockDim.x + threadIdx.x;
  if (gid >= K * 384) return;
  int k = gid / 384, n = gid - k * 384;
  WT[gid] = Wih[(size_t)n * K + k];
}

// One block per batch lane; 128 threads = gate slots; 500-step recurrence.
// GI holds precomputed x@Wih^T + bih for all timesteps (WMMA GEMM).
__global__ void k_gru(const float* __restrict__ GI, const float* __restrict__ Whh,
                      const float* __restrict__ bhh, float* __restrict__ out,
                      int L, int reverse, int outStride, int outOffset) {
  int b = blockIdx.x, n = threadIdx.x;
  __shared__ float h[HG];
  h[n] = 0.0f; __syncthreads();
  const float* Wr = Whh + (size_t)n * HG;
  const float* Wz = Whh + (size_t)(HG + n) * HG;
  const float* Wn = Whh + (size_t)(2 * HG + n) * HG;
  const float br = bhh[n], bz = bhh[HG + n], bn_ = bhh[2 * HG + n];
  for (int s = 0; s < L; ++s) {
    int t = reverse ? (L - 1 - s) : s;
    size_t row = (size_t)b * L + t;
    float gr = br, gz = bz, gn = bn_;
#pragma unroll 8
    for (int k = 0; k < HG; ++k) {
      float hk = h[k];
      gr += Wr[k] * hk; gz += Wz[k] * hk; gn += Wn[k] * hk;
    }
    const float* gi = GI + row * 384;
    float r  = 1.0f / (1.0f + __expf(-(gi[n] + gr)));
    float z  = 1.0f / (1.0f + __expf(-(gi[HG + n] + gz)));
    float nn = tanhf(gi[2 * HG + n] + r * gn);
    float hn2 = (1.0f - z) * nn + z * h[n];
    __syncthreads();
    h[n] = hn2;
    out[row * outStride + outOffset + n] = hn2;
    __syncthreads();
  }
}

__global__ void k_ln(const float* __restrict__ z, const float* __restrict__ g,
                     const float* __restrict__ b, float* __restrict__ out, int M, int D) {
  int wv = (blockIdx.x * blockDim.x + threadIdx.x) >> 5;
  int lane = threadIdx.x & 31;
  if (wv >= M) return;
  const float* r = z + (size_t)wv * D;
  float s = 0.f, ss = 0.f;
  for (int i = lane; i < D; i += 32) { float v = r[i]; s += v; ss += v * v; }
#pragma unroll
  for (int m = 16; m >= 1; m >>= 1) { s += __shfl_xor(s, m, 32); ss += __shfl_xor(ss, m, 32); }
  float mean = s / (float)D;
  float var  = ss / (float)D - mean * mean;
  float inv  = rsqrtf(var + 1e-5f);
  for (int i = lane; i < D; i += 32)
    out[(size_t)wv * D + i] = g[i] * (r[i] - mean) * inv + b[i];
}

// ---------------------------------------------------------------------------
extern "C" void kernel_launch(void* const* d_in, const int* in_sizes, int n_in,
                              void* d_out, int out_size, void* d_ws, size_t ws_size,
                              hipStream_t stream) {
  const float* x           = (const float*)d_in[0];
  const int*   edge_index  = (const int*)d_in[1];   // [NT,2,E_TYPE]
  const int*   onset_index = (const int*)d_in[2];   // [2,E_ON]
  const int*   onset_idx   = (const int*)d_in[3];   // [N_POOL]
  // d_in[4] = lengths (equal, unused)
  const float* Wnbr[3]  = {(const float*)d_in[5], (const float*)d_in[6], (const float*)d_in[7]};
  const float* Wroot[3] = {(const float*)d_in[8], (const float*)d_in[9], (const float*)d_in[10]};
  const float* bl[3]    = {(const float*)d_in[11], (const float*)d_in[12], (const float*)d_in[13]};
  const float* bn1_b = (const float*)d_in[14]; const float* bn1_g = (const float*)d_in[15];
  const float* bn2_b = (const float*)d_in[16]; const float* bn2_g = (const float*)d_in[17];
  const float* emb_W = (const float*)d_in[18]; const float* emb_b = (const float*)d_in[19];
  // gru leaves: 20 + l*8 + d*4 + {Wih,Whh,bih,bhh}
  const float* ln_b = (const float*)d_in[36]; const float* ln_g = (const float*)d_in[37];
  const float* pitch_emb = (const float*)d_in[38];
  const float* proj1_W = (const float*)d_in[39]; const float* proj1_b = (const float*)d_in[40];
  const float* proj2_W = (const float*)d_in[41]; const float* proj2_b = (const float*)d_in[42];
  const float* spell_emb = (const float*)d_in[43];
  const float* trans_W = (const float*)d_in[44]; const float* trans_b = (const float*)d_in[45];
  float* out = (float*)d_out;

  // --- workspace carve-up (floats) ---
  float* base = (float*)d_ws; size_t off = 0;
  auto alloc = [&](size_t n) { float* p = base + off; off += n; return p; };
  float* hA     = alloc((size_t)N_NODES * 256);  // current h / hp / GI
  float* hB     = alloc((size_t)N_NODES * 256);  // acc / ht / z1 / seqA
  float* mbuf   = alloc((size_t)N_NODES * 256);  // seg-sums / pooled / z2 / seqB
  float* invcnt = alloc((size_t)NT * N_NODES);
  float* cntOn  = alloc(N_NODES);
  float* wsum   = alloc(256 * 256);
  float* bsum   = alloc(256);
  float* wT     = alloc(256 * 384);
  float* wpad   = alloc(KP1 * 256);
  float* stats  = alloc(512);
  (void)ws_size; (void)n_in; (void)in_sizes; (void)out_size;

  auto gemm = [&](int NO, const float* A, int lda, const float* rs,
                  const float* W, float* C, int M, int K) {
    int blocks = ((M / 16) * (NO / 64) + 7) / 8;   // 8 waves / 256-thread block
    if (NO == 256)      k_gemm_bf16<256><<<blocks, 256, 0, stream>>>(A, lda, rs, W, C, M, K);
    else if (NO == 384) k_gemm_bf16<384><<<blocks, 256, 0, stream>>>(A, lda, rs, W, C, M, K);
    else                k_gemm_bf16<128><<<blocks, 256, 0, stream>>>(A, lda, rs, W, C, M, K);
  };
  auto grid = [](long long n) { return (unsigned)((n + 255) / 256); };

  // 1) embedding -> hA [N,64]
  k_embed<<<grid((long long)N_NODES * 64), 256, 0, stream>>>(x, emb_W, emb_b,
                                                             pitch_emb, spell_emb, hA);
  // 2) per-type in-degree counts (layer invariant) -> invcnt = 1/max(cnt,1)
  hipMemsetAsync(invcnt, 0, (size_t)NT * N_NODES * sizeof(float), stream);
  k_edge_count<<<grid((long long)NT * E_TYPE), 256, 0, stream>>>(edge_index, invcnt);
  k_invert_cnt<<<grid((long long)NT * N_NODES), 256, 0, stream>>>(invcnt, NT * N_NODES);

  // 3) hetero SAGE layers: acc = h@ΣWroot + Σ_t mean_t@Wnbr[t] + Σb, /7, relu
  int din = 64;
  for (int l = 0; l < 3; ++l) {
    int sz = din * 256;
    k_sum_types<<<grid(sz), 256, 0, stream>>>(Wroot[l], wsum, sz);
    k_sum_types<<<grid(256), 256, 0, stream>>>(bl[l], bsum, 256);
    k_fill_rowvec<<<grid((long long)N_NODES * 256), 256, 0, stream>>>(hB, bsum, N_NODES, 256);
    gemm(256, hA, din, nullptr, wsum, hB, N_NODES, din);
    for (int t = 0; t < NT; ++t) {
      hipMemsetAsync(mbuf, 0, (size_t)N_NODES * din * sizeof(float), stream);
      const int* src = edge_index + (size_t)t * 2 * E_TYPE;
      const int* dst = src + E_TYPE;
      k_scatter4<<<grid((long long)E_TYPE * (din / 4)), 256, 0, stream>>>(src, dst, hA,
                                                                          mbuf, E_TYPE, din);
      gemm(256, mbuf, din, invcnt + (size_t)t * N_NODES,
           Wnbr[l] + (size_t)t * sz, hB, N_NODES, din);
    }
    k_scale_relu<<<grid((long long)N_NODES * 256), 256, 0, stream>>>(hB, hA,
                                                                     N_NODES * 256, 1.0f / 7.0f);
    din = 256;
  }
  k_rownorm<<<grid((long long)N_NODES * 32), 256, 0, stream>>>(hA, N_NODES, 256);

  // 4) ht = h @ trans_W + trans_b -> hB
  k_fill_rowvec<<<grid((long long)N_NODES * 256), 256, 0, stream>>>(hB, trans_b, N_NODES, 256);
  gemm(256, hA, 256, nullptr, trans_W, hB, N_NODES, 256);

  // 5) onset pooling: pooled = ht (self loops) + scatter(ht[src] by dst); gather idx
  hipMemcpyAsync(mbuf, hB, (size_t)N_NODES * 256 * sizeof(float),
                 hipMemcpyDeviceToDevice, stream);
  k_scatter4<<<grid((long long)E_ON * 64), 256, 0, stream>>>(onset_index, onset_index + E_ON,
                                                             hB, mbuf, E_ON, 256);
  k_fill_const<<<grid(N_NODES), 256, 0, stream>>>(cntOn, 1.0f, N_NODES);
  k_count_on<<<grid(E_ON), 256, 0, stream>>>(onset_index + E_ON, cntOn, E_ON);
  k_build_hp<<<grid((long long)N_POOL * KP1), 256, 0, stream>>>(mbuf, cntOn, onset_idx, x, hA);

  // 6) proj1 (K padded 257->288) + relu + BN1 -> hB
  k_pad_w1<<<grid((long long)KP1 * 256), 256, 0, stream>>>(proj1_W, wpad);
  k_fill_rowvec<<<grid((long long)N_POOL * 256), 256, 0, stream>>>(hB, proj1_b, N_POOL, 256);
  gemm(256, hA, KP1, nullptr, wpad, hB, N_POOL, KP1);
  k_relu<<<grid((long long)N_POOL * 256), 256, 0, stream>>>(hB, N_POOL * 256);
  k_colstats<<<256, 256, 0, stream>>>(hB, N_POOL, 256, stats);
  k_bn_apply<<<grid((long long)N_POOL * 256), 256, 0, stream>>>(hB, stats, bn1_g, bn1_b,
                                                                N_POOL, 256);
  // 7) proj2 + relu + BN2 -> mbuf [50k,128]
  k_fill_rowvec<<<grid((long long)N_POOL * 128), 256, 0, stream>>>(mbuf, proj2_b, N_POOL, 128);
  gemm(128, hB, 256, nullptr, proj2_W, mbuf, N_POOL, 256);
  k_relu<<<grid((long long)N_POOL * 128), 256, 0, stream>>>(mbuf, N_POOL * 128);
  k_colstats<<<128, 256, 0, stream>>>(mbuf, N_POOL, 128, stats);
  k_bn_apply<<<grid((long long)N_POOL * 128), 256, 0, stream>>>(mbuf, stats, bn2_g, bn2_b,
                                                                N_POOL, 128);

  // 8) bidirectional GRU x2: GI = X @ Wih^T + bih hoisted into WMMA GEMM,
  //    recurrence parallel over 100 batch lanes.
  const float* seqIn = mbuf; int K0 = 128; float* seqOut = hB;   // layer 0: mbuf -> hB
  for (int l = 0; l < 2; ++l) {
    for (int d = 0; d < 2; ++d) {
      const float* Wih = (const float*)d_in[20 + l * 8 + d * 4 + 0];
      const float* Whh = (const float*)d_in[20 + l * 8 + d * 4 + 1];
      const float* bih = (const float*)d_in[20 + l * 8 + d * 4 + 2];
      const float* bhh = (const float*)d_in[20 + l * 8 + d * 4 + 3];
      k_transpose<<<grid((long long)K0 * 384), 256, 0, stream>>>(Wih, wT, K0);
      k_fill_rowvec<<<grid((long long)N_POOL * 384), 256, 0, stream>>>(hA, bih, N_POOL, 384);
      gemm(384, seqIn, K0, nullptr, wT, hA, N_POOL, K0);
      k_gru<<<BATCH, HG, 0, stream>>>(hA, Whh, bhh, seqOut, SEQL, d, 256, d * HG);
    }
    seqIn = hB; K0 = 256; seqOut = mbuf;   // layer 1: hB -> mbuf
  }

  // 9) LayerNorm -> d_out [50k, 256] fp32
  k_ln<<<grid((long long)N_POOL * 32), 256, 0, stream>>>(mbuf, ln_g, ln_b, out, N_POOL, 256);
}
